// GroupedQueryAttention_4964982194242
// MI455X (gfx1250) — matrix operand order repaired and checked by reading, not compiled, not run on hardware
//
#include <hip/hip_runtime.h>

// ---------------------------------------------------------------------------
// GQA block for MI455X (gfx1250): bf16 WMMA GEMMs + flash attention.
// B=2, S=2048, HID=2048, NH=16, NKV=4, D=128. Matrix math on
// v_wmma_f32_16x16x32_bf16 (wave32); tile feeds via async global->LDS copies
// (GLOBAL_LOAD_ASYNC_TO_LDS_B128, ASYNCcnt) with double-buffered LDS.
// ---------------------------------------------------------------------------

typedef __attribute__((ext_vector_type(16))) __bf16 bf16x16;
typedef __attribute__((ext_vector_type(8)))  float  v8f;
typedef __attribute__((ext_vector_type(4)))  unsigned int u32x4;
typedef int v4i_b128 __attribute__((vector_size(16)));   // matches builtin pointee

#define WMMA_BF16(A, B, C) \
  __builtin_amdgcn_wmma_f32_16x16x32_bf16(false, (A), false, (B), (short)0, (C), false, false)

// ---- async global->LDS plumbing (gfx1250) ---------------------------------
#if defined(__has_builtin)
#if __has_builtin(__builtin_amdgcn_global_load_async_to_lds_b128)
#define ASYNC_LDS 1
#endif
#if __has_builtin(__builtin_amdgcn_s_wait_asynccnt)
#define HAVE_WAIT_ASYNC 1
#endif
#endif
#ifndef ASYNC_LDS
#define ASYNC_LDS 0
#endif

// 16-byte global -> LDS copy; async (no VGPR bounce) when available.
__device__ __forceinline__ void cp16(void* lds_dst, const void* gsrc) {
#if ASYNC_LDS
  __builtin_amdgcn_global_load_async_to_lds_b128(
      (__attribute__((address_space(1))) v4i_b128*)gsrc,
      (__attribute__((address_space(3))) v4i_b128*)lds_dst, 0, 0);
#else
  *(u32x4*)lds_dst = *(const u32x4*)gsrc;
#endif
}

__device__ __forceinline__ void wait_async_le4() {
#if ASYNC_LDS
#if defined(HAVE_WAIT_ASYNC)
  __builtin_amdgcn_s_wait_asynccnt(4);
#else
  asm volatile("s_wait_asynccnt 0x4" ::: "memory");
#endif
#endif
}
__device__ __forceinline__ void wait_async0() {
#if ASYNC_LDS
#if defined(HAVE_WAIT_ASYNC)
  __builtin_amdgcn_s_wait_asynccnt(0);
#else
  asm volatile("s_wait_asynccnt 0x0" ::: "memory");
#endif
#endif
}
// ---------------------------------------------------------------------------

__device__ __forceinline__ v8f vzero8() {
  v8f z = {0.f, 0.f, 0.f, 0.f, 0.f, 0.f, 0.f, 0.f};
  return z;
}

__device__ __forceinline__ unsigned short f2bf(float f) {
  union { float f; unsigned int u; } v;
  v.f = f;
  unsigned int lsb = (v.u >> 16) & 1u;
  v.u += 0x7fffu + lsb;              // round-to-nearest-even
  return (unsigned short)(v.u >> 16);
}

// A-fragment (16x32 bf16): lo 16B at p, hi 16B at p+32 (per ISA VGPR layout).
__device__ __forceinline__ bf16x16 ldsA_frag(const char* p) {
  union { u32x4 q[2]; bf16x16 v; } u;
  u.q[0] = *(const u32x4*)(p);
  u.q[1] = *(const u32x4*)(p + 32);
  return u.v;
}
// B-fragment (32x16 bf16) from an [n][k] LDS layout: lo at p, hi at p+32.
__device__ __forceinline__ bf16x16 ldsB_frag(const char* p) {
  union { u32x4 q[2]; bf16x16 v; } u;
  u.q[0] = *(const u32x4*)(p);
  u.q[1] = *(const u32x4*)(p + 32);
  return u.v;
}

// ---------------------------------------------------------------------------
// Elementwise conversion kernels
// ---------------------------------------------------------------------------
__global__ __launch_bounds__(256) void f32_to_bf16_k(
    const float* __restrict__ s, unsigned short* __restrict__ d, int n) {
  int i = blockIdx.x * 256 + threadIdx.x;
  if (i < n) d[i] = f2bf(s[i]);
}

// W [K][N] f32 -> WT [N][K] bf16
__global__ __launch_bounds__(256) void transpose_bf16_k(
    const float* __restrict__ src, unsigned short* __restrict__ dst, int K, int N) {
  int idx = blockIdx.x * 256 + threadIdx.x;
  if (idx >= K * N) return;
  int k = idx % K;
  int n = idx / K;
  dst[(size_t)n * K + k] = f2bf(src[(size_t)k * N + n]);
}

// RoPE + relayout: src f32 [B*S][NHp*128] -> dst bf16 [B][NHp][S][128]
__global__ __launch_bounds__(256) void rope_k(
    const float* __restrict__ src, unsigned short* __restrict__ dst, int NHp) {
  constexpr int S = 2048, D = 128, B = 2;
  int total = B * S * NHp * (D / 2);
  int idx = blockIdx.x * 256 + threadIdx.x;
  if (idx >= total) return;
  int d2 = idx & 63;
  int h  = (idx >> 6) % NHp;
  int s  = (idx / (64 * NHp)) % S;
  int b  = idx / (64 * NHp * S);
  int rowLen = NHp * D;
  const float* p = src + ((size_t)b * S + s) * rowLen + h * D + 2 * d2;
  float xr = p[0], xi = p[1];
  float inv = __expf(-(float)(2 * d2) * (9.210340371976184f / 128.0f)); // theta^(-2i/128)
  float ang = (float)s * inv;
  float sn, cs;
  __sincosf(ang, &sn, &cs);
  float rr = xr * cs - xi * sn;
  float ri = xr * sn + xi * cs;
  unsigned short* q = dst + (((size_t)b * NHp + h) * S + s) * D + 2 * d2;
  q[0] = f2bf(rr);
  q[1] = f2bf(ri);
}

// V relayout: f32 [B*S][512] -> bf16 [B][NKV][S][128]
__global__ __launch_bounds__(256) void v_reorder_k(
    const float* __restrict__ src, unsigned short* __restrict__ dst) {
  constexpr int S = 2048, D = 128, B = 2, NKV = 4;
  int idx = blockIdx.x * 256 + threadIdx.x;
  if (idx >= B * S * NKV * D) return;
  int d = idx & 127;
  int h = (idx >> 7) & 3;
  int s = (idx >> 9) & 2047;
  int b = idx >> 20;
  dst[(((size_t)b * NKV + h) * S + s) * D + d] =
      f2bf(src[((size_t)b * S + s) * (NKV * D) + h * D + d]);
}

// ---------------------------------------------------------------------------
// GEMM: C[M][N] f32 = A[M][K] bf16 @ W, with W given transposed BT[N][K] bf16.
// Block tile 128x128, 256 threads (8 waves as 4x2), wave tile 32x64, K-step 32.
// Double-buffered LDS, async prefetch of tile t+1 while tile t feeds WMMA.
// LDS pitch 80 B: (20*m mod 64) distinct for m=0..15 -> conflict-free frags.
// ---------------------------------------------------------------------------
__global__ __launch_bounds__(256) void gemm_bf16_f32(
    const unsigned short* __restrict__ A,
    const unsigned short* __restrict__ BT,
    float* __restrict__ C,
    int K, int lda, int ldb, int ldc) {
  constexpr int TBUF = 2 * 128 * 80;  // A tile + B tile per buffer (20480 B)
  __shared__ __align__(16) char smem[2 * TBUF];

  const int t    = threadIdx.x;
  const int lane = t & 31;
  const int wave = t >> 5;
  const int wm   = (wave & 3) * 32;
  const int wn   = (wave >> 2) * 64;
  const int mBase = blockIdx.y * 128;
  const int nBase = blockIdx.x * 128;
  const int rA  = lane & 15;
  const int sel = lane >> 4;

  v8f acc[2][4];
#pragma unroll
  for (int i = 0; i < 2; ++i)
#pragma unroll
    for (int j = 0; j < 4; ++j) acc[i][j] = vzero8();

  const int rowL   = t >> 2;        // 0..63
  const int colE   = (t & 3) * 8;   // element offset
  const int colB16 = (t & 3) * 16;  // byte offset

  auto issue_tile = [&](int k0, int buf) {
    char* dA = smem + buf * TBUF;
    char* dB = dA + 128 * 80;
#pragma unroll
    for (int i = 0; i < 2; ++i) {
      int row = rowL + i * 64;
      cp16(dA + row * 80 + colB16, A  + (size_t)(mBase + row) * lda + k0 + colE);
      cp16(dB + row * 80 + colB16, BT + (size_t)(nBase + row) * ldb + k0 + colE);
    }
  };

  issue_tile(0, 0);
  const int T = K >> 5;
  for (int ti = 0; ti < T; ++ti) {
    if (ti + 1 < T) {
      issue_tile((ti + 1) << 5, (ti + 1) & 1);  // prefetch next tile
      wait_async_le4();                          // own tile-ti copies done
    } else {
      wait_async0();
    }
    __syncthreads();                             // everyone's tile ti visible

    const char* sA = smem + (ti & 1) * TBUF;
    const char* sB = sA + 128 * 80;

    bf16x16 aF[2], bF[4];
#pragma unroll
    for (int i = 0; i < 2; ++i)
      aF[i] = ldsA_frag(sA + (wm + i * 16 + rA) * 80 + sel * 16);
#pragma unroll
    for (int j = 0; j < 4; ++j)
      bF[j] = ldsB_frag(sB + (wn + j * 16 + rA) * 80 + sel * 16);

#pragma unroll
    for (int i = 0; i < 2; ++i)
#pragma unroll
      for (int j = 0; j < 4; ++j)
        acc[i][j] = WMMA_BF16(aF[i], bF[j], acc[i][j]);
    __syncthreads();                             // reads done before reuse
  }

  // C/D layout: n = lane&15 ; m = (lane>>4)*8 + vgpr
#pragma unroll
  for (int i = 0; i < 2; ++i)
#pragma unroll
    for (int j = 0; j < 4; ++j)
#pragma unroll
      for (int r = 0; r < 8; ++r) {
        int gm = mBase + wm + i * 16 + sel * 8 + r;
        int gn = nBase + wn + j * 16 + rA;
        C[(size_t)gm * ldc + gn] = acc[i][j][r];
      }
}

// ---------------------------------------------------------------------------
// Flash attention: one block = (b, h, 64 q rows); 4 waves x 16 q rows each.
// K-tiles of 64. Online softmax; P re-shaped via per-wave LDS scratch.
// Q/K tiles arrive via async global->LDS; V is transposed through VGPRs.
// ---------------------------------------------------------------------------
__global__ __launch_bounds__(128) void flash_attn_k(
    const unsigned short* __restrict__ Qr,
    const unsigned short* __restrict__ Kr,
    const unsigned short* __restrict__ Vb,
    float* __restrict__ Ctx) {
  constexpr int S = 2048, D = 128, NH = 16, NKV = 4, HID = 2048;
  constexpr int QP = 272, VP = 144, PP = 144;  // LDS row pitches (16B-aligned, skewed)
  __shared__ __align__(16) char lds[64 * QP + 64 * QP + 128 * VP + 4 * 16 * PP]; // 62464 B
  char* sQ  = lds;
  char* sK  = lds + 64 * QP;
  char* sVt = sK + 64 * QP;
  char* sP  = sVt + 128 * VP;

  const int t    = threadIdx.x;
  const int lane = t & 31;
  const int wave = t >> 5;
  const int rA   = lane & 15;
  const int sel  = lane >> 4;
  const int b = blockIdx.z, h = blockIdx.y;
  const int kvh = h >> 2;                       // groups = NH/NKV = 4
  const int qBase = blockIdx.x * 64;
  const float scale = 0.08838834764831845f;     // 1/sqrt(128)

  const unsigned short* Qg = Qr + ((size_t)(b * NH + h) * S + qBase) * D;
  const unsigned short* Kg = Kr + (size_t)(b * NKV + kvh) * S * D;
  const unsigned short* Vg = Vb + (size_t)(b * NKV + kvh) * S * D;

  // Q tile: 64 rows x 128 bf16 (async copy; completed by first tile wait)
#pragma unroll
  for (int i = 0; i < 8; ++i) {
    int idx = t + i * 128;
    int row = idx >> 4, ch = idx & 15;
    cp16(sQ + row * QP + ch * 16, Qg + (size_t)row * D + ch * 8);
  }

  v8f oacc[8];
#pragma unroll
  for (int dt = 0; dt < 8; ++dt) oacc[dt] = vzero8();
  float mrow[8], lrow[8];
#pragma unroll
  for (int j = 0; j < 8; ++j) { mrow[j] = -3.0e38f; lrow[j] = 0.f; }

  char* sPw = sP + wave * (16 * PP);

  for (int kt = 0; kt < S / 64; ++kt) {
    const int kBase = kt * 64;
    // K tile [kcol][d] — already the [n][k] layout for the scores B operand
#pragma unroll
    for (int i = 0; i < 8; ++i) {
      int idx = t + i * 128;
      int row = idx >> 4, ch = idx & 15;
      cp16(sK + row * QP + ch * 16, Kg + (size_t)(kBase + row) * D + ch * 8);
    }
    // V tile transposed into [d][kcol] for the PV B operand
#pragma unroll
    for (int i = 0; i < 8; ++i) {
      int idx = t + i * 128;
      int kcol = idx >> 4, dbase = (idx & 15) * 8;
      union { u32x4 q; unsigned short u[8]; } vv;
      vv.q = *(const u32x4*)(Vg + (size_t)(kBase + kcol) * D + dbase);
#pragma unroll
      for (int e = 0; e < 8; ++e)
        *(unsigned short*)(sVt + (dbase + e) * VP + kcol * 2) = vv.u[e];
    }
    wait_async0();
    __syncthreads();

    // scores = Q @ K^T : per wave 16x64, K-dim = d (4 chunks of 32)
    bf16x16 aF[4];
    const char* qp = sQ + (wave * 16 + rA) * QP + sel * 16;
#pragma unroll
    for (int c = 0; c < 4; ++c) aF[c] = ldsA_frag(qp + c * 64);

    v8f sacc[4];
#pragma unroll
    for (int nt = 0; nt < 4; ++nt) {
      sacc[nt] = vzero8();
      const char* kp = sK + (nt * 16 + rA) * QP + sel * 16;
#pragma unroll
      for (int c = 0; c < 4; ++c) {
        bf16x16 bF = ldsB_frag(kp + c * 64);
        sacc[nt] = WMMA_BF16(aF[c], bF, sacc[nt]);
      }
    }

#pragma unroll
    for (int nt = 0; nt < 4; ++nt)
#pragma unroll
      for (int j = 0; j < 8; ++j) sacc[nt][j] *= scale;

    // row max over 64 cols: local over nt, then across the 16-lane half-wave
    float vmax[8];
#pragma unroll
    for (int j = 0; j < 8; ++j)
      vmax[j] = fmaxf(fmaxf(sacc[0][j], sacc[1][j]), fmaxf(sacc[2][j], sacc[3][j]));
#pragma unroll
    for (int off = 1; off < 16; off <<= 1)
#pragma unroll
      for (int j = 0; j < 8; ++j)
        vmax[j] = fmaxf(vmax[j], __shfl_xor(vmax[j], off, 32));

    float alpha[8], rsum[8];
#pragma unroll
    for (int j = 0; j < 8; ++j) {
      float mnew = fmaxf(mrow[j], vmax[j]);
      alpha[j] = __expf(mrow[j] - mnew);
      mrow[j] = mnew;
      rsum[j] = 0.f;
    }
#pragma unroll
    for (int nt = 0; nt < 4; ++nt)
#pragma unroll
      for (int j = 0; j < 8; ++j) {
        float p = __expf(sacc[nt][j] - mrow[j]);
        sacc[nt][j] = p;
        rsum[j] += p;
      }
#pragma unroll
    for (int off = 1; off < 16; off <<= 1)
#pragma unroll
      for (int j = 0; j < 8; ++j)
        rsum[j] += __shfl_xor(rsum[j], off, 32);
#pragma unroll
    for (int j = 0; j < 8; ++j) lrow[j] = lrow[j] * alpha[j] + rsum[j];
#pragma unroll
    for (int dt = 0; dt < 8; ++dt)
#pragma unroll
      for (int j = 0; j < 8; ++j) oacc[dt][j] *= alpha[j];

    // P (C-layout) -> per-wave LDS as bf16 [16 q][64 kcol] so it can be
    // re-fetched in A-fragment layout
#pragma unroll
    for (int nt = 0; nt < 4; ++nt)
#pragma unroll
      for (int j = 0; j < 8; ++j)
        *(unsigned short*)(sPw + (sel * 8 + j) * PP + (nt * 16 + rA) * 2) =
            f2bf(sacc[nt][j]);
    asm volatile("s_wait_dscnt 0" ::: "memory");   // intra-wave LDS RAW fence

    // O += P @ V : K-dim = kcol (2 chunks of 32), N = d (8 tiles of 16)
#pragma unroll
    for (int c = 0; c < 2; ++c) {
      bf16x16 aP = ldsA_frag(sPw + rA * PP + sel * 16 + c * 64);
#pragma unroll
      for (int dt = 0; dt < 8; ++dt) {
        bf16x16 bV = ldsB_frag(sVt + (dt * 16 + rA) * VP + sel * 16 + c * 64);
        oacc[dt] = WMMA_BF16(aP, bV, oacc[dt]);
      }
    }
    __syncthreads();
  }

  float inv[8];
#pragma unroll
  for (int j = 0; j < 8; ++j) inv[j] = 1.0f / lrow[j];
#pragma unroll
  for (int dt = 0; dt < 8; ++dt)
#pragma unroll
    for (int j = 0; j < 8; ++j) {
      int row = qBase + wave * 16 + sel * 8 + j;
      int col = h * D + dt * 16 + rA;
      Ctx[((size_t)b * S + row) * HID + col] = oacc[dt][j] * inv[j];
    }
}

// ---------------------------------------------------------------------------
// Host orchestration
// ---------------------------------------------------------------------------
extern "C" void kernel_launch(void* const* d_in, const int* in_sizes, int n_in,
                              void* d_out, int out_size, void* d_ws, size_t ws_size,
                              hipStream_t stream) {
  constexpr int B = 2, S = 2048, H = 2048, NH = 16, NKV = 4, D = 128, KVD = 512;
  constexpr int M = B * S;  // 4096

  const float* X  = (const float*)d_in[0];
  const float* Wq = (const float*)d_in[1];
  const float* Wk = (const float*)d_in[2];
  const float* Wv = (const float*)d_in[3];
  const float* Wo = (const float*)d_in[4];
  float* out = (float*)d_out;
  char* ws = (char*)d_ws;

  // workspace layout (bytes); Ctx/Ctxb alias dead Q/K/V f32 buffers
  constexpr size_t off_Xb   = 0;                                // 16 MB
  constexpr size_t off_WqT  = off_Xb   + (size_t)M * H * 2;     // 8 MB
  constexpr size_t off_WkT  = off_WqT  + (size_t)H * H * 2;     // 2 MB
  constexpr size_t off_WvT  = off_WkT  + (size_t)H * KVD * 2;   // 2 MB
  constexpr size_t off_WoT  = off_WvT  + (size_t)H * KVD * 2;   // 8 MB
  constexpr size_t off_Qf   = off_WoT  + (size_t)H * H * 2;     // 32 MB (f32)
  constexpr size_t off_Kf   = off_Qf   + (size_t)M * H * 4;     // 8 MB
  constexpr size_t off_Vf   = off_Kf   + (size_t)M * KVD * 4;   // 8 MB
  constexpr size_t off_Qr   = off_Vf   + (size_t)M * KVD * 4;   // 16 MB bf16
  constexpr size_t off_Kr   = off_Qr   + (size_t)M * H * 2;     // 4 MB
  constexpr size_t off_Vb   = off_Kr   + (size_t)M * KVD * 2;   // 4 MB
  constexpr size_t off_Ctx  = off_Qf;                           // reuse Qf
  constexpr size_t off_Ctxb = off_Kf;                           // reuse Kf+Vf

  unsigned short* Xb   = (unsigned short*)(ws + off_Xb);
  unsigned short* WqT  = (unsigned short*)(ws + off_WqT);
  unsigned short* WkT  = (unsigned short*)(ws + off_WkT);
  unsigned short* WvT  = (unsigned short*)(ws + off_WvT);
  unsigned short* WoT  = (unsigned short*)(ws + off_WoT);
  float*          Qf   = (float*)(ws + off_Qf);
  float*          Kf   = (float*)(ws + off_Kf);
  float*          Vf   = (float*)(ws + off_Vf);
  unsigned short* Qr   = (unsigned short*)(ws + off_Qr);
  unsigned short* Kr   = (unsigned short*)(ws + off_Kr);
  unsigned short* Vb   = (unsigned short*)(ws + off_Vb);
  float*          Ctx  = (float*)(ws + off_Ctx);
  unsigned short* Ctxb = (unsigned short*)(ws + off_Ctxb);

  // 1) precision conversion / weight transposition
  {
    int n = M * H;
    f32_to_bf16_k<<<(n + 255) / 256, 256, 0, stream>>>(X, Xb, n);
  }
  transpose_bf16_k<<<(H * H   + 255) / 256, 256, 0, stream>>>(Wq, WqT, H, H);
  transpose_bf16_k<<<(H * KVD + 255) / 256, 256, 0, stream>>>(Wk, WkT, H, KVD);
  transpose_bf16_k<<<(H * KVD + 255) / 256, 256, 0, stream>>>(Wv, WvT, H, KVD);
  transpose_bf16_k<<<(H * H   + 255) / 256, 256, 0, stream>>>(Wo, WoT, H, H);

  // 2) QKV projections (bf16 WMMA, f32 accumulate)
  gemm_bf16_f32<<<dim3(H / 128,   M / 128), 256, 0, stream>>>(Xb, WqT, Qf, H, H, H, H);
  gemm_bf16_f32<<<dim3(KVD / 128, M / 128), 256, 0, stream>>>(Xb, WkT, Kf, H, H, H, KVD);
  gemm_bf16_f32<<<dim3(KVD / 128, M / 128), 256, 0, stream>>>(Xb, WvT, Vf, H, H, H, KVD);

  // 3) RoPE (f32 math) + relayout to [b][h][s][d] bf16
  {
    int nq = B * S * NH  * (D / 2);
    int nk = B * S * NKV * (D / 2);
    rope_k<<<(nq + 255) / 256, 256, 0, stream>>>(Qf, Qr, NH);
    rope_k<<<(nk + 255) / 256, 256, 0, stream>>>(Kf, Kr, NKV);
    int nv = B * S * NKV * D;
    v_reorder_k<<<(nv + 255) / 256, 256, 0, stream>>>(Vf, Vb);
  }

  // 4) flash attention -> context f32 [B*S][H]
  flash_attn_k<<<dim3(S / 64, NH, B), 128, 0, stream>>>(Qr, Kr, Vb, Ctx);

  // 5) output projection
  {
    int n = M * H;
    f32_to_bf16_k<<<(n + 255) / 256, 256, 0, stream>>>(Ctx, Ctxb, n);
  }
  gemm_bf16_f32<<<dim3(H / 128, M / 128), 256, 0, stream>>>(Ctxb, WoT, out, H, H, H, H);
}
